// Scaled_Dot_Product_Attention_21741124452466
// MI455X (gfx1250) — matrix operand order repaired and checked by reading, not compiled, not run on hardware
//
#include <hip/hip_runtime.h>

typedef __attribute__((ext_vector_type(16))) _Float16 v16h;
typedef __attribute__((ext_vector_type(8)))  float    v8f;
typedef __attribute__((ext_vector_type(2)))  _Float16 h2;
typedef __attribute__((ext_vector_type(4)))  _Float16 h4;
typedef __attribute__((ext_vector_type(8)))  _Float16 h8;

#define S_LEN 2048
#define DHEAD 64
#define BM    128            // Q rows per workgroup (8 waves x 16)
#define BN    64             // keys per LDS tile
#define NWAVE 8
#define NT    (S_LEN / BN)   // 32 tiles

#define SCALE_LOG2 0.18033688011112042f   // (1/sqrt(64)) * log2(e), folded into Q

// 16-lane xor-reduction (single DPP instruction per stage) + running-max combine in asm
// (avoids llvm canonicalize ops on asm outputs)
__device__ __forceinline__ float rowmax16c(float x, float mold) {
  float t, r;
  asm("v_max_num_f32_dpp %0, %1, %1 row_xmask:1 row_mask:0xf bank_mask:0xf bound_ctrl:1" : "=v"(t) : "v"(x));
  asm("v_max_num_f32_dpp %0, %1, %1 row_xmask:2 row_mask:0xf bank_mask:0xf bound_ctrl:1" : "=v"(x) : "v"(t));
  asm("v_max_num_f32_dpp %0, %1, %1 row_xmask:4 row_mask:0xf bank_mask:0xf bound_ctrl:1" : "=v"(t) : "v"(x));
  asm("v_max_num_f32_dpp %0, %1, %1 row_xmask:8 row_mask:0xf bank_mask:0xf bound_ctrl:1" : "=v"(x) : "v"(t));
  asm("v_max_num_f32_e32 %0, %1, %2" : "=v"(r) : "v"(x), "v"(mold));
  return r;
}
__device__ __forceinline__ float rowsum16(float x) {
  float t;
  asm("v_add_f32_dpp %0, %1, %1 row_xmask:1 row_mask:0xf bank_mask:0xf bound_ctrl:1" : "=v"(t) : "v"(x));
  asm("v_add_f32_dpp %0, %1, %1 row_xmask:2 row_mask:0xf bank_mask:0xf bound_ctrl:1" : "=v"(x) : "v"(t));
  asm("v_add_f32_dpp %0, %1, %1 row_xmask:4 row_mask:0xf bank_mask:0xf bound_ctrl:1" : "=v"(t) : "v"(x));
  asm("v_add_f32_dpp %0, %1, %1 row_xmask:8 row_mask:0xf bank_mask:0xf bound_ctrl:1" : "=v"(x) : "v"(t));
  return x;
}

__global__ __launch_bounds__(256)
void fa_fwd_f16wmma(const float* __restrict__ Q, const float* __restrict__ K,
                    const float* __restrict__ V, float* __restrict__ O)
{
  // Key order inside a tile is PERMUTED for the PV matmul's K dimension:
  //   k' = 4*(key % 16) + key/16   (so the 4 score subtiles per row are contiguous in P)
  // P (A operand) and Vt (B operand) both use k'; WMMA reduction is order-invariant.
  __shared__ __align__(16) _Float16 Kt[2][BN * DHEAD];   // [key][d]    f16 (natural order)
  __shared__ __align__(16) _Float16 Vt[2][DHEAD * BN];   // [d][k']     f16 (transposed+permuted)
  __shared__ __align__(16) _Float16 Pt[NWAVE][16 * BN];  // per-wave P  [row][k']

  const int tid  = threadIdx.x;
  const int wave = tid >> 5;
  const int lane = tid & 31;
  const int l16  = lane & 15;
  const int hi   = lane >> 4;            // 0: lanes 0-15, 1: lanes 16-31

  const int bh = blockIdx.y;             // 0..63
  const size_t hoff = (size_t)bh * S_LEN * DHEAD;
  const float* Qh = Q + hoff;
  const float* Kh = K + hoff;
  const float* Vh = V + hoff;
  float*       Oh = O + hoff;

  const int qbase = blockIdx.x * BM + wave * 16;

  // ---- Q A-fragments: 16x32 f16 x2, pre-scaled into the exp2 domain ----
  v16h aq[2];
  {
    const float* qr = Qh + (size_t)(qbase + l16) * DHEAD;
    const int koff = hi * 16;
    #pragma unroll
    for (int f = 0; f < 2; ++f) {
      #pragma unroll
      for (int v = 0; v < 8; ++v) {
        const int k = f * 32 + 2 * v + koff;
        const float2 t = *(const float2*)(qr + k);
        aq[f][2 * v]     = (_Float16)(t.x * SCALE_LOG2);
        aq[f][2 * v + 1] = (_Float16)(t.y * SCALE_LOG2);
      }
    }
  }

  // ---- cooperative K/V tile staging (prefetch regs -> LDS f16, double-buffered) ----
  const int kkey = tid >> 2;             // 0..63 (key within tile), natural order
  const int kd   = (tid & 3) * 16;       // 16 consecutive d floats per thread
  const int vd   = tid & 63;             // d for V
  const int vib  = (tid >> 6) * 4;       // i-base (key%16 group), 4 i's per thread

  float kf[16], vf[16];
  auto load_tile_regs = [&](int it) {
    const int kb = it * BN;
    const float4* kp = (const float4*)(Kh + (size_t)(kb + kkey) * DHEAD + kd);
    #pragma unroll
    for (int c = 0; c < 4; ++c) {
      const float4 a = kp[c];
      kf[4*c+0]=a.x; kf[4*c+1]=a.y; kf[4*c+2]=a.z; kf[4*c+3]=a.w;
    }
    // vf[ii*4+g] = V[key = 16*g + (vib+ii)][vd]  -> lands at k' = 4*(vib+ii)+g
    #pragma unroll
    for (int ii = 0; ii < 4; ++ii)
      #pragma unroll
      for (int g = 0; g < 4; ++g)
        vf[ii*4+g] = Vh[(size_t)(kb + 16*g + vib + ii) * DHEAD + vd];
  };
  auto store_tile_lds = [&](int buf) {
    #pragma unroll
    for (int c = 0; c < 2; ++c) {
      h8 kk;
      #pragma unroll
      for (int i = 0; i < 8; ++i) kk[i] = (_Float16)kf[8*c+i];
      *(h8*)&Kt[buf][kkey * DHEAD + kd + 8*c] = kk;
    }
    #pragma unroll
    for (int ii = 0; ii < 4; ++ii) {
      h4 w;
      #pragma unroll
      for (int g = 0; g < 4; ++g) w[g] = (_Float16)vf[ii*4+g];
      *(h4*)&Vt[buf][vd * BN + 4*(vib+ii)] = w;      // [d][k']
    }
  };

  // ---- softmax state + output accumulators ----
  float m[8], l[8];
  v8f o[4];
  #pragma unroll
  for (int j = 0; j < 8; ++j) { m[j] = -__builtin_inff(); l[j] = 0.0f; }
  #pragma unroll
  for (int dt = 0; dt < 4; ++dt) { v8f z = {}; o[dt] = z; }

  load_tile_regs(0);
  store_tile_lds(0);
  __syncthreads();

  for (int it = 0; it < NT; ++it) {
    const int cur = it & 1;
    if (it + 1 < NT) load_tile_regs(it + 1);   // global prefetch hidden under compute

    // ---- S = Q K^T : four 16-key subtiles, K-dim 64 via 2 WMMA each ----
    v8f sc[4];
    #pragma unroll
    for (int st = 0; st < 4; ++st) {
      v16h bk0, bk1;
      const _Float16* kp = &Kt[cur][(st * 16 + l16) * DHEAD + hi * 16];
      #pragma unroll
      for (int v = 0; v < 8; ++v) {
        const h2 t0 = *(const h2*)(kp + 2 * v);
        const h2 t1 = *(const h2*)(kp + 32 + 2 * v);
        bk0[2*v] = t0[0]; bk0[2*v+1] = t0[1];
        bk1[2*v] = t1[0]; bk1[2*v+1] = t1[1];
      }
      v8f c = {};
      c = __builtin_amdgcn_wmma_f32_16x16x32_f16(false, aq[0], false, bk0, (short)0, c, false, false);
      c = __builtin_amdgcn_wmma_f32_16x16x32_f16(false, aq[1], false, bk1, (short)0, c, false, false);
      sc[st] = c;
    }

    // ---- online softmax (exp2 domain; one pass per 64 keys) ----
    v8f cv;
    #pragma unroll
    for (int j = 0; j < 8; ++j) {
      const float s0 = sc[0][j], s1 = sc[1][j], s2 = sc[2][j], s3 = sc[3][j];
      const float mn = rowmax16c(fmaxf(fmaxf(s0, s1), fmaxf(s2, s3)), m[j]);
      const float corr = __builtin_amdgcn_exp2f(m[j] - mn);
      m[j] = mn;
      cv[j] = corr;
      const float e0 = __builtin_amdgcn_exp2f(s0 - mn);
      const float e1 = __builtin_amdgcn_exp2f(s1 - mn);
      const float e2 = __builtin_amdgcn_exp2f(s2 - mn);
      const float e3 = __builtin_amdgcn_exp2f(s3 - mn);
      l[j] = l[j] * corr + rowsum16((e0 + e1) + (e2 + e3));
      h4 ph;
      ph[0] = (_Float16)e0; ph[1] = (_Float16)e1;
      ph[2] = (_Float16)e2; ph[3] = (_Float16)e3;
      *(h4*)&Pt[wave][(j + hi * 8) * BN + l16 * 4] = ph;   // k' = 4*l16 + st, contiguous
    }
    #pragma unroll
    for (int dt = 0; dt < 4; ++dt) o[dt] *= cv;   // vector rescale -> v_pk_mul_f32

    // ---- P A-fragments (16x32 x2, K' order) ----
    v16h ap[2];
    #pragma unroll
    for (int c = 0; c < 2; ++c) {
      const int kadd = c * 32 + hi * 16;
      #pragma unroll
      for (int v = 0; v < 8; ++v) {
        const int k = 2 * v + kadd;
        const h2 t = *(const h2*)&Pt[wave][l16 * BN + k];
        ap[c][2*v] = t[0]; ap[c][2*v+1] = t[1];
      }
    }

    // ---- O += P V  (4 d-subtiles of 16 columns, 2 k'-chunks each) ----
    #pragma unroll
    for (int dt = 0; dt < 4; ++dt) {
      v16h bv0, bv1;
      const _Float16* vp = &Vt[cur][(dt * 16 + l16) * BN + hi * 16];
      #pragma unroll
      for (int v = 0; v < 8; ++v) {
        const h2 t0 = *(const h2*)(vp + 2 * v);
        const h2 t1 = *(const h2*)(vp + 32 + 2 * v);
        bv0[2*v] = t0[0]; bv0[2*v+1] = t0[1];
        bv1[2*v] = t1[0]; bv1[2*v+1] = t1[1];
      }
      o[dt] = __builtin_amdgcn_wmma_f32_16x16x32_f16(false, ap[0], false, bv0, (short)0, o[dt], false, false);
      o[dt] = __builtin_amdgcn_wmma_f32_16x16x32_f16(false, ap[1], false, bv1, (short)0, o[dt], false, false);
    }

    if (it + 1 < NT) store_tile_lds((it + 1) & 1);  // other buffer; safe pre-barrier
    __syncthreads();
  }

  // ---- epilogue: O / l ----
  float invl[8];
  #pragma unroll
  for (int j = 0; j < 8; ++j) invl[j] = 1.0f / l[j];
  float* orow = Oh + (size_t)qbase * DHEAD;
  #pragma unroll
  for (int dt = 0; dt < 4; ++dt) {
    #pragma unroll
    for (int j = 0; j < 8; ++j) {
      const int r = j + hi * 8;
      orow[(size_t)r * DHEAD + dt * 16 + l16] = o[dt][j] * invl[j];
    }
  }
}

extern "C" void kernel_launch(void* const* d_in, const int* in_sizes, int n_in,
                              void* d_out, int out_size, void* d_ws, size_t ws_size,
                              hipStream_t stream) {
  const float* q = (const float*)d_in[0];
  const float* k = (const float*)d_in[1];
  const float* v = (const float*)d_in[2];
  float* out = (float*)d_out;
  dim3 grid(S_LEN / BM, 4 * 16);   // 16 q-tiles x 64 (B*H) heads
  fa_fwd_f16wmma<<<grid, 256, 0, stream>>>(q, k, v, out);
}
